// PoseGNN_78658031059124
// MI455X (gfx1250) — matrix operand order repaired and checked by reading, not compiled, not run on hardware
//
#include <hip/hip_runtime.h>
#include <hip/hip_bf16.h>

// ---------------------------------------------------------------------------
// PoseGNN (3-layer GAT) for MI455X / gfx1250, wave32.
//   GEMM0  : V_WMMA_F32_16X16X4_F32   (K=4 exactly, f32 inputs)
//   GEMM1/2: V_WMMA_F32_16X16X32_BF16 with the 64x512 bf16 B-panel staged
//            into LDS by the Tensor Data Mover (tensor_load_to_lds +
//            s_wait_tensorcnt), B fragments read via ds_load_b128.
//   Edge phase: 3-pass segment softmax with f32 global atomics
//   BatchNorm: 2-pass column stats, apply fused with f32->bf16 conversion
// Per-column bias before BN is a mathematical no-op (BN removes column mean),
// so b0/b1/b2 are intentionally unused.
// ---------------------------------------------------------------------------

#define NN 20000
#define EE 240000
#define ET (EE + NN)          // edges + self loops
#define HH 4
#define CC 128
#define DD 512                // H*C
#define EPS_BN 1e-5f
#define NEG_SLOPE 0.2f

typedef __attribute__((ext_vector_type(16))) __bf16 v16bf;
typedef __attribute__((ext_vector_type(8)))  __bf16 v8bf;
typedef __attribute__((ext_vector_type(8)))  float  v8f;
typedef __attribute__((ext_vector_type(2)))  float  v2f;
typedef __attribute__((ext_vector_type(4)))  unsigned u32x4;
typedef __attribute__((ext_vector_type(8)))  unsigned u32x8;

// ---------------------------------------------------------------------------
// Layer 0 GEMM: out[N,512] = x[N,4] @ W0[4,512], one V_WMMA_F32_16X16X4_F32
// per 16x16 output tile (K=4 consumed in a single instruction).
// ---------------------------------------------------------------------------
__global__ void gemm0_f32_wmma(const float* __restrict__ x,
                               const float* __restrict__ W0,
                               float* __restrict__ out) {
  int wid = (blockIdx.x * blockDim.x + threadIdx.x) >> 5;
  const int ntiles = DD / 16;                 // 32
  const int total  = (NN / 16) * ntiles;      // 40000 waves
  if (wid >= total) return;
  int mt = wid / ntiles, nt = wid % ntiles;
  int m0 = mt * 16, n0 = nt * 16;
  int lane = threadIdx.x & 31;
  int r = lane & 15, half = lane >> 4;

  // A 16x4 f32: lanes 0-15 hold K=0,1; lanes 16-31 hold K=2,3 (row M = lane&15)
  const float* ap = x + (size_t)(m0 + r) * 4 + half * 2;
  v2f a; a[0] = ap[0]; a[1] = ap[1];
  // B 4x16 f32: lanes 0-15 hold K=0,1; lanes 16-31 hold K=2,3 (col N = lane&15)
  v2f b; b[0] = W0[(half * 2 + 0) * DD + n0 + r];
         b[1] = W0[(half * 2 + 1) * DD + n0 + r];
  v8f c = {};
  c = __builtin_amdgcn_wmma_f32_16x16x4_f32(false, a, false, b,
                                            (short)0, c, false, false);
#pragma unroll
  for (int v = 0; v < 8; ++v)
    out[(size_t)(m0 + v + 8 * half) * DD + n0 + r] = c[v];
}

// ---------------------------------------------------------------------------
// bf16 GEMM with TDM-staged B panel.
// Block = 256 threads (8 waves) computes 128 rows x 64 cols.
// The 64x512 bf16 panel of Wt (64 KB) is DMA'd into LDS once by wave 0.
// ---------------------------------------------------------------------------
__device__ __forceinline__ v16bf ld16(const __bf16* p0, const __bf16* p1) {
  v8bf lo = *(const v8bf*)p0;
  v8bf hi = *(const v8bf*)p1;
  v16bf r;
#pragma unroll
  for (int i = 0; i < 8; ++i) { r[i] = lo[i]; r[i + 8] = hi[i]; }
  return r;
}

__global__ void gemm_bf16_tdm(const __bf16* __restrict__ A,
                              const __bf16* __restrict__ Wt,
                              float* __restrict__ out,
                              int Din, int Dout) {
  // Sole LDS object in this kernel -> LDS offset 0 (used by the D#).
  __shared__ __bf16 lb[64 * DD];              // 64 KB panel: [64 cols][Din K]

  int ngrp = Dout / 64;
  int bm = blockIdx.x / ngrp, ng = blockIdx.x % ngrp;
  int n0 = ng * 64;
  int w = threadIdx.x >> 5, lane = threadIdx.x & 31;
  int m0 = bm * 128 + w * 16;

  // ---- Tensor DMA descriptor (CDNA5 ISA) -----------
  // View the panel as 8-byte elements: row = Din/4 elems, 64 rows.
  unsigned long long ga = (unsigned long long)(Wt + (size_t)n0 * Din);
  unsigned rowElems = (unsigned)(Din / 4);    // 128 for Din=512
  u32x4 g0;
  g0[0] = 1u;                                 // count=1, user mode
  g0[1] = 0u;                                 // lds_addr = 0
  g0[2] = (unsigned)(ga & 0xffffffffu);       // global_addr[31:0]
  g0[3] = (unsigned)((ga >> 32) & 0x1ffffffu) // global_addr[56:32]
          | (2u << 30);                       // type = 2 ("image")
  u32x8 g1;
  g1[0] = (3u << 16);                         // data_size = 3 (8 bytes)
  g1[1] = (rowElems << 16);                   // tensor_dim0[15:0] @ bits 63:48
  g1[2] = (64u << 16);                        // tensor_dim0 hi=0; tensor_dim1=64
  g1[3] = (rowElems << 16);                   // tensor_dim1 hi=0; tile_dim0=128
  g1[4] = 64u;                                // tile_dim1=64; tile_dim2=0
  g1[5] = rowElems;                           // tensor_dim0_stride[31:0]
  g1[6] = 0u;                                 // stride hi; tensor_dim1_stride lo
  g1[7] = 0u;

  if (w == 0) {
    asm volatile("tensor_load_to_lds %0, %1"
                 :: "s"(g0), "s"(g1) : "memory");
    __builtin_amdgcn_s_wait_tensorcnt(0);
  }
  __syncthreads();                            // publish LDS panel to all waves

  if (m0 < NN) {
    int r = lane & 15, half = lane >> 4;
    v8f acc[4] = {};
    const __bf16* arow = A + (size_t)(m0 + r) * Din;
    for (int k0 = 0; k0 < Din; k0 += 32) {
      // A 16x32 bf16: per-lane K sets {half*8..+7} and {16+half*8..+7}
      v16bf a = ld16(arow + k0 + half * 8, arow + k0 + 16 + half * 8);
#pragma unroll
      for (int t = 0; t < 4; ++t) {
        // B 32x16 bf16 from LDS: K sets {half*8..+7} and {16+half*8..+7} (2x ds_load_b128)
        const __bf16* bp = lb + (size_t)(t * 16 + r) * Din + k0 + half * 8;
        v16bf b = ld16(bp, bp + 16);
        acc[t] = __builtin_amdgcn_wmma_f32_16x16x32_bf16(false, a, false, b,
                                                         (short)0, acc[t],
                                                         false, false);
      }
    }
#pragma unroll
    for (int t = 0; t < 4; ++t)
#pragma unroll
      for (int v = 0; v < 8; ++v)
        out[(size_t)(m0 + v + 8 * half) * Dout + n0 + t * 16 + r] = acc[t][v];
  }
}

// W[Din,Dout] f32 (row-major) -> Wt[Dout,Din] bf16 (K contiguous per column)
__global__ void transpose_to_bf16(const float* __restrict__ W,
                                  __bf16* __restrict__ Wt,
                                  int Din, int Dout) {
  int i = blockIdx.x * blockDim.x + threadIdx.x;
  if (i >= Din * Dout) return;
  int k = i / Dout, n = i % Dout;
  Wt[(size_t)n * Din + k] = (__bf16)W[i];
}

// ---------------------------------------------------------------------------
// a_s[n,h] = sum_c hW[n,h*C+c]*att_s[h,c]   (one wave per (node,head))
// ---------------------------------------------------------------------------
__global__ void att_kernel(const float* __restrict__ hw,
                           const float* __restrict__ atts,
                           const float* __restrict__ attd,
                           float* __restrict__ as_, float* __restrict__ ad_,
                           int heads, int Dout) {
  int n = blockIdx.x;
  int h = threadIdx.x >> 5, lane = threadIdx.x & 31;
  const float* row = hw + (size_t)n * Dout + h * CC;
  const float* ws  = atts + h * CC;
  const float* wd  = attd + h * CC;
  float ss = 0.f, sd = 0.f;
#pragma unroll
  for (int c = lane; c < CC; c += 32) {
    float v = row[c];
    ss += v * ws[c];
    sd += v * wd[c];
  }
#pragma unroll
  for (int off = 16; off; off >>= 1) {
    ss += __shfl_xor(ss, off, 32);
    sd += __shfl_xor(sd, off, 32);
  }
  if (lane == 0) { as_[n * heads + h] = ss; ad_[n * heads + h] = sd; }
}

// ---------------------------------------------------------------------------
// Edge phase: segment softmax over dst + message scatter.
// ---------------------------------------------------------------------------
__device__ __forceinline__ void edge_sd(const long long* ei, int e,
                                        int& s, int& d) {
  if (e < EE) { s = (int)ei[e]; d = (int)ei[EE + e]; }
  else        { s = e - EE; d = s; }   // self loop
}
// order-preserving float<->uint map for atomicMax on signed floats
__device__ __forceinline__ unsigned fkey(float f) {
  unsigned u = __float_as_uint(f);
  return (u & 0x80000000u) ? ~u : (u | 0x80000000u);
}
__device__ __forceinline__ float funkey(unsigned u) {
  return (u & 0x80000000u) ? __uint_as_float(u & 0x7fffffffu)
                           : __uint_as_float(~u);
}

__global__ void edge_logits_max(const long long* __restrict__ ei,
                                const float* __restrict__ as_,
                                const float* __restrict__ ad_,
                                float* __restrict__ logits,
                                unsigned* __restrict__ mkey, int heads) {
  int e = blockIdx.x * blockDim.x + threadIdx.x;
  if (e >= ET) return;
  int s, d; edge_sd(ei, e, s, d);
  for (int h = 0; h < heads; ++h) {
    float l = as_[s * heads + h] + ad_[d * heads + h];
    l = l > 0.f ? l : NEG_SLOPE * l;
    logits[(size_t)e * heads + h] = l;
    atomicMax(&mkey[d * heads + h], fkey(l));
  }
}

__global__ void edge_exp_sum(const long long* __restrict__ ei,
                             const float* __restrict__ logits,
                             const unsigned* __restrict__ mkey,
                             float* __restrict__ eexp,
                             float* __restrict__ ssum, int heads) {
  int e = blockIdx.x * blockDim.x + threadIdx.x;
  if (e >= ET) return;
  int s, d; edge_sd(ei, e, s, d);
  (void)s;
  for (int h = 0; h < heads; ++h) {
    float m  = funkey(mkey[d * heads + h]);
    float ex = __expf(logits[(size_t)e * heads + h] - m);
    eexp[(size_t)e * heads + h] = ex;
    atomicAdd(&ssum[d * heads + h], ex);
  }
}

// one wave per edge: each lane handles C/32 floats, 32 apart, per head
__global__ void edge_scatter(const long long* __restrict__ ei,
                             const float* __restrict__ hw,
                             const float* __restrict__ eexp,
                             const float* __restrict__ ssum,
                             float* __restrict__ agg, int heads, int Dout) {
  int wid = (blockIdx.x * blockDim.x + threadIdx.x) >> 5;
  if (wid >= ET) return;
  int lane = threadIdx.x & 31;
  int s, d; edge_sd(ei, wid, s, d);
  for (int h = 0; h < heads; ++h) {
    float alpha = eexp[(size_t)wid * heads + h] /
                  (ssum[d * heads + h] + 1e-16f);
    const float* src = hw  + (size_t)s * Dout + h * CC;
    float*       dst = agg + (size_t)d * Dout + h * CC;
    for (int c = lane; c < CC; c += 32)
      atomicAdd(&dst[c], src[c] * alpha);
  }
}

// ---------------------------------------------------------------------------
// BatchNorm (training stats, biased var) + ReLU + bf16 conversion
// ---------------------------------------------------------------------------
__global__ void bn_stats(const float* __restrict__ xin,
                         float* __restrict__ colsum,
                         float* __restrict__ colsq, int Dout) {
  float s[2] = {0.f, 0.f}, q[2] = {0.f, 0.f};
  for (int row = blockIdx.x; row < NN; row += gridDim.x) {
    const float* r = xin + (size_t)row * Dout;
#pragma unroll
    for (int cc = 0; cc < 2; ++cc) {
      int col = threadIdx.x + cc * 256;
      if (col < Dout) { float v = r[col]; s[cc] += v; q[cc] += v * v; }
    }
  }
#pragma unroll
  for (int cc = 0; cc < 2; ++cc) {
    int col = threadIdx.x + cc * 256;
    if (col < Dout) {
      atomicAdd(&colsum[col], s[cc]);
      atomicAdd(&colsq[col],  q[cc]);
    }
  }
}

__global__ void bn_finalize(const float* __restrict__ colsum,
                            const float* __restrict__ colsq,
                            const float* __restrict__ gamma,
                            const float* __restrict__ beta,
                            float* __restrict__ scale,
                            float* __restrict__ shift, int Dout) {
  int c = blockIdx.x * blockDim.x + threadIdx.x;
  if (c >= Dout) return;
  float mu  = colsum[c] * (1.f / NN);
  float var = colsq[c] * (1.f / NN) - mu * mu;
  float sc  = gamma[c] * rsqrtf(var + EPS_BN);
  scale[c] = sc;
  shift[c] = beta[c] - mu * sc;
}

__global__ void bn_apply(const float* __restrict__ xin,
                         const float* __restrict__ scale,
                         const float* __restrict__ shift,
                         __bf16* __restrict__ outbf,
                         float* __restrict__ outf, int Dout, int relu) {
  size_t i = (size_t)blockIdx.x * blockDim.x + threadIdx.x;
  size_t total = (size_t)NN * Dout;
  if (i >= total) return;
  int col = (int)(i % Dout);
  float v = xin[i] * scale[col] + shift[col];
  if (relu) v = v > 0.f ? v : 0.f;
  if (outbf) outbf[i] = (__bf16)v;
  if (outf)  outf[i]  = v;
}

// ---------------------------------------------------------------------------
extern "C" void kernel_launch(void* const* d_in, const int* in_sizes, int n_in,
                              void* d_out, int out_size, void* d_ws,
                              size_t ws_size, hipStream_t stream) {
  (void)in_sizes; (void)n_in; (void)out_size; (void)ws_size;
  const float*     x   = (const float*)d_in[0];
  const long long* ei  = (const long long*)d_in[1];
  const float*     W0  = (const float*)d_in[2];
  const float*     as0 = (const float*)d_in[3];
  const float*     ad0 = (const float*)d_in[4];
  const float*     g0  = (const float*)d_in[6];
  const float*     be0 = (const float*)d_in[7];
  const float*     W1  = (const float*)d_in[8];
  const float*     as1 = (const float*)d_in[9];
  const float*     ad1 = (const float*)d_in[10];
  const float*     g1  = (const float*)d_in[12];
  const float*     be1 = (const float*)d_in[13];
  const float*     W2  = (const float*)d_in[14];
  const float*     as2 = (const float*)d_in[15];
  const float*     ad2 = (const float*)d_in[16];
  const float*     g2  = (const float*)d_in[18];
  const float*     be2 = (const float*)d_in[19];

  // workspace carving (256B aligned)
  char* ws = (char*)d_ws;
  size_t off = 0;
  auto carve = [&](size_t bytes) -> char* {
    char* p = ws + off;
    off += (bytes + 255) & ~(size_t)255;
    return p;
  };
  float*    HW  = (float*)carve((size_t)NN * DD * 4);   // GEMM output h
  float*    AGG = (float*)carve((size_t)NN * DD * 4);   // aggregated messages
  __bf16*   ABF = (__bf16*)carve((size_t)NN * DD * 2);  // bf16 activations
  __bf16*   WT  = (__bf16*)carve((size_t)DD * DD * 2);  // transposed weights
  float*    AS  = (float*)carve((size_t)NN * HH * 4);
  float*    AD  = (float*)carve((size_t)NN * HH * 4);
  unsigned* MK  = (unsigned*)carve((size_t)NN * HH * 4);
  float*    SS  = (float*)carve((size_t)NN * HH * 4);
  float*    LG  = (float*)carve((size_t)ET * HH * 4);
  float*    EX  = (float*)carve((size_t)ET * HH * 4);
  float*    CS  = (float*)carve(DD * 4);
  float*    CQ  = (float*)carve(DD * 4);
  float*    SC  = (float*)carve(DD * 4);
  float*    SH  = (float*)carve(DD * 4);

  auto run_edges_bn = [&](int heads, int Dout, const float* atts,
                          const float* attd, const float* gamma,
                          const float* beta, __bf16* nextbf, float* outf,
                          int relu) {
    att_kernel<<<NN, heads * 32, 0, stream>>>(HW, atts, attd, AS, AD,
                                              heads, Dout);
    hipMemsetAsync(AGG, 0, (size_t)NN * Dout * 4, stream);
    hipMemsetAsync(MK,  0, (size_t)NN * heads * 4, stream);
    hipMemsetAsync(SS,  0, (size_t)NN * heads * 4, stream);
    hipMemsetAsync(CS,  0, (size_t)Dout * 4, stream);
    hipMemsetAsync(CQ,  0, (size_t)Dout * 4, stream);
    int eb = (ET + 255) / 256;
    edge_logits_max<<<eb, 256, 0, stream>>>(ei, AS, AD, LG, MK, heads);
    edge_exp_sum<<<eb, 256, 0, stream>>>(ei, LG, MK, EX, SS, heads);
    int sb = (int)(((size_t)ET * 32 + 255) / 256);
    edge_scatter<<<sb, 256, 0, stream>>>(ei, HW, EX, SS, AGG, heads, Dout);
    bn_stats<<<512, 256, 0, stream>>>(AGG, CS, CQ, Dout);
    bn_finalize<<<(Dout + 255) / 256, 256, 0, stream>>>(CS, CQ, gamma, beta,
                                                        SC, SH, Dout);
    size_t tot = (size_t)NN * Dout;
    bn_apply<<<(int)((tot + 255) / 256), 256, 0, stream>>>(AGG, SC, SH,
                                                           nextbf, outf,
                                                           Dout, relu);
  };

  const int MB = (NN + 127) / 128;            // 157 row-blocks of 128 rows

  // ---- layer 0: in=4 -> 4 heads x 128, concat
  {
    int waves = (NN / 16) * (DD / 16);
    gemm0_f32_wmma<<<(waves * 32 + 255) / 256, 256, 0, stream>>>(x, W0, HW);
    run_edges_bn(HH, DD, as0, ad0, g0, be0, ABF, nullptr, 1);
  }
  // ---- layer 1: 512 -> 4 heads x 128, concat
  {
    transpose_to_bf16<<<(DD * DD + 255) / 256, 256, 0, stream>>>(W1, WT,
                                                                 DD, DD);
    gemm_bf16_tdm<<<MB * (DD / 64), 256, 0, stream>>>(ABF, WT, HW, DD, DD);
    run_edges_bn(HH, DD, as1, ad1, g1, be1, ABF, nullptr, 1);
  }
  // ---- layer 2: 512 -> 1 head x 128, mean(=identity for H=1), no ReLU
  {
    transpose_to_bf16<<<(DD * CC + 255) / 256, 256, 0, stream>>>(W2, WT,
                                                                 DD, CC);
    gemm_bf16_tdm<<<MB * (CC / 64), 256, 0, stream>>>(ABF, WT, HW, DD, CC);
    run_edges_bn(1, CC, as2, ad2, g2, be2, nullptr, (float*)d_out, 0);
  }
}
